// GatedLightningIndexer_50019189129746
// MI455X (gfx1250) — matrix operand order repaired and checked by reading, not compiled, not run on hardware
//
#include <hip/hip_runtime.h>

#define B_    2
#define T_    4096
#define HID_  1024
#define DI_   64
#define NH_   4
#define SCALE_ 0.125f

typedef __attribute__((ext_vector_type(16))) __bf16 v16bf;
typedef __attribute__((ext_vector_type(8)))  float  v8f;

union BF16x16 { v16bf v; unsigned short u[16]; uint4 q[2]; };
union F8      { v8f v; float f[8]; };

// round-to-nearest-even f32 -> bf16 (bit manipulation; no backend surprises)
__device__ __forceinline__ unsigned short f2bf(float f) {
  unsigned int u = __float_as_uint(f);
  u += 0x7FFFu + ((u >> 16) & 1u);
  return (unsigned short)(u >> 16);
}

// ---------------------------------------------------------------------------
// Kernel 0: one-shot f32 -> bf16 conversion (memory bound, runs once per src)
// n must be a multiple of 8.
// ---------------------------------------------------------------------------
__global__ __launch_bounds__(256) void cvt_bf16_kernel(
    const float* __restrict__ src, unsigned short* __restrict__ dst, int n)
{
  int i = (blockIdx.x * blockDim.x + threadIdx.x) * 8;
  if (i >= n) return;
  float4 a = *(const float4*)(src + i);
  float4 b = *(const float4*)(src + i + 4);
  union { uint4 q; unsigned short u[8]; } o;
  o.u[0] = f2bf(a.x); o.u[1] = f2bf(a.y); o.u[2] = f2bf(a.z); o.u[3] = f2bf(a.w);
  o.u[4] = f2bf(b.x); o.u[5] = f2bf(b.y); o.u[6] = f2bf(b.z); o.u[7] = f2bf(b.w);
  *(uint4*)(dst + i) = o.q;
}

// ---------------------------------------------------------------------------
// Kernel 1: fused q/k projection from pre-converted bf16 operands.
// Logical output [8192, 320] bf16 (wqk = [Wq rows 0..255 | Wk rows 0..63]):
//   cols 0..255   -> qb [B*T, 256]
//   cols 256..319 -> kb [B*T, 64]
// One wave computes 16 rows x 64 cols; block = 8 waves = 128 rows.
// Pure uint4 bf16 loads + v_wmma_f32_16x16x32_bf16; no per-iteration cvt.
// ---------------------------------------------------------------------------
__global__ __launch_bounds__(256) void proj_qk_kernel(
    const unsigned short* __restrict__ xb, const unsigned short* __restrict__ wqk,
    unsigned short* __restrict__ qb, unsigned short* __restrict__ kb)
{
  const int lane = threadIdx.x & 31;
  const int wave = threadIdx.x >> 5;
  const int rowbase = blockIdx.y * 128 + wave * 16;
  const int colbase = blockIdx.x * 64;
  const int lm = lane & 15, lh = lane >> 4;

  const unsigned short* xrow = xb + (size_t)(rowbase + lm) * HID_;
  const unsigned short* wrow[4];
#pragma unroll
  for (int ct = 0; ct < 4; ++ct)
    wrow[ct] = wqk + (size_t)(colbase + ct * 16 + lm) * HID_;

  F8 acc[4];
#pragma unroll
  for (int ct = 0; ct < 4; ++ct)
#pragma unroll
    for (int i = 0; i < 8; ++i) acc[ct].f[i] = 0.f;

  for (int k0 = 0; k0 < HID_; k0 += 32) {
    // A tile (16x32 bf16), ISA 16-bit A layout: per lane two contiguous
    // 8-element chunks at K offsets (lane/16)*8 and +16.
    BF16x16 a;
    const unsigned short* ap = xrow + k0 + lh * 8;
    a.q[0] = *(const uint4*)(ap);
    a.q[1] = *(const uint4*)(ap + 16);
#pragma unroll
    for (int ct = 0; ct < 4; ++ct) {
      // B tile (32x16 bf16): lane = column, same K chunks as A: (lane/16)*8 and +16
      BF16x16 bm;
      const unsigned short* bp = wrow[ct] + k0 + lh * 8;
      bm.q[0] = *(const uint4*)(bp);
      bm.q[1] = *(const uint4*)(bp + 16);
      acc[ct].v = __builtin_amdgcn_wmma_f32_16x16x32_bf16(
          false, a.v, false, bm.v, (short)0, acc[ct].v, false, false);
    }
  }

  // C/D layout: VGPR i -> M = i + 8*(lane/16), N = lane%16
#pragma unroll
  for (int ct = 0; ct < 4; ++ct)
#pragma unroll
    for (int i = 0; i < 8; ++i) {
      int r = rowbase + i + lh * 8;
      int c = colbase + ct * 16 + lm;
      unsigned short h = f2bf(acc[ct].f[i]);
      if (c < 256) qb[(size_t)r * 256 + c] = h;
      else         kb[(size_t)r * 64 + (c - 256)] = h;
    }
}

// ---------------------------------------------------------------------------
// Kernel 2: w projection + sigmoid gate (tiny: 32768 dots of length 1024)
// ---------------------------------------------------------------------------
__global__ __launch_bounds__(256) void proj_w_kernel(
    const float* __restrict__ x, const float* __restrict__ Ww,
    const float* __restrict__ bw, float* __restrict__ swg)
{
  int tid = blockIdx.x * blockDim.x + threadIdx.x;   // 0..32767
  int row = tid >> 2, h = tid & 3;
  const float4* xr = (const float4*)(x + (size_t)row * HID_);
  const float4* wr = (const float4*)(Ww + (size_t)h * HID_);
  float s = 0.f;
#pragma unroll 4
  for (int i = 0; i < HID_ / 4; ++i) {
    float4 a = xr[i], b = wr[i];
    s += a.x * b.x + a.y * b.y + a.z * b.z + a.w * b.w;
  }
  s += bw[h];
  swg[tid] = 1.0f / (1.0f + __expf(-s));
}

// ---------------------------------------------------------------------------
// Kernel 3: gated causal scores (store-bandwidth-bound: 134 MB f32 out).
// Block = 8 waves; wave computes a 16(t) x 64(s) out tile for one batch.
// Per head: two chained v_wmma_f32_16x16x32_bf16 (K = 64).
// ---------------------------------------------------------------------------
__global__ __launch_bounds__(256) void score_kernel(
    const unsigned short* __restrict__ qb, const unsigned short* __restrict__ kb,
    const float* __restrict__ swg, const float* __restrict__ idx_bias,
    float* __restrict__ out)
{
  const int lane = threadIdx.x & 31;
  const int wave = threadIdx.x >> 5;
  const int b = blockIdx.z;
  const int rowbase = blockIdx.y * 128 + wave * 16;  // t
  const int colbase = blockIdx.x * 64;               // s
  const int lm = lane & 15, lh = lane >> 4;

  float* outb = out + (size_t)b * T_ * T_;
  const float NEGINF = -__builtin_inff();

  if (colbase > rowbase + 15) {   // fully above diagonal: s > t everywhere
#pragma unroll
    for (int ct = 0; ct < 4; ++ct)
#pragma unroll
      for (int i = 0; i < 8; ++i) {
        int t = rowbase + i + lh * 8;
        int s = colbase + ct * 16 + lm;
        outb[(size_t)t * T_ + s] = NEGINF;
      }
    return;
  }

  // A operands: q rows for 4 heads, two K-steps of 32 (ISA 16-bit A layout)
  BF16x16 A[NH_][2];
  const unsigned short* qrow =
      qb + (size_t)(b * T_ + rowbase + lm) * (NH_ * DI_);
#pragma unroll
  for (int h = 0; h < NH_; ++h)
#pragma unroll
    for (int ks = 0; ks < 2; ++ks) {
      const unsigned short* p = qrow + h * DI_ + ks * 32 + lh * 8;
      A[h][ks].q[0] = *(const uint4*)(p);
      A[h][ks].q[1] = *(const uint4*)(p + 16);
    }

  // gate values: sigmoid(w)[b, t_row, 0..3] per handled row (f32x4, h-contig)
  float4 g[8];
#pragma unroll
  for (int i = 0; i < 8; ++i)
    g[i] = *(const float4*)(swg + (size_t)(b * T_ + rowbase + i + lh * 8) * NH_);

  float ib[NH_];
#pragma unroll
  for (int h = 0; h < NH_; ++h) ib[h] = idx_bias[h];

#pragma unroll
  for (int ct = 0; ct < 4; ++ct) {
    // B operands: k rows as columns; lane = column (s), same K chunks as A
    BF16x16 Bm[2];
    const unsigned short* krow =
        kb + (size_t)(b * T_ + colbase + ct * 16 + lm) * DI_;
#pragma unroll
    for (int ks = 0; ks < 2; ++ks) {
      const unsigned short* p = krow + ks * 32 + lh * 8;
      Bm[ks].q[0] = *(const uint4*)(p);
      Bm[ks].q[1] = *(const uint4*)(p + 16);
    }

    F8 acc;
#pragma unroll
    for (int i = 0; i < 8; ++i) acc.f[i] = 0.f;

#pragma unroll
    for (int h = 0; h < NH_; ++h) {
      v8f z = {};
      F8 sc;
      sc.v = __builtin_amdgcn_wmma_f32_16x16x32_bf16(
          false, A[h][0].v, false, Bm[0].v, (short)0, z, false, false);
      sc.v = __builtin_amdgcn_wmma_f32_16x16x32_bf16(
          false, A[h][1].v, false, Bm[1].v, (short)0, sc.v, false, false);
      float bias = ib[h];
#pragma unroll
      for (int i = 0; i < 8; ++i) {
        float zz = sc.f[i] * SCALE_ + bias;
        float sg = 1.0f / (1.0f + __expf(-zz));
        float gv = (h == 0) ? g[i].x : (h == 1) ? g[i].y
                 : (h == 2) ? g[i].z : g[i].w;
        acc.f[i] += gv * sg;
      }
    }

#pragma unroll
    for (int i = 0; i < 8; ++i) {
      int t = rowbase + i + lh * 8;
      int s = colbase + ct * 16 + lm;
      outb[(size_t)t * T_ + s] = (s > t) ? NEGINF : acc.f[i];
    }
  }
}

// ---------------------------------------------------------------------------
extern "C" void kernel_launch(void* const* d_in, const int* in_sizes, int n_in,
                              void* d_out, int out_size, void* d_ws, size_t ws_size,
                              hipStream_t stream) {
  const float* x  = (const float*)d_in[0];
  const float* Wq = (const float*)d_in[1];
  const float* Wk = (const float*)d_in[2];
  const float* Ww = (const float*)d_in[3];
  const float* bw = (const float*)d_in[4];
  const float* ib = (const float*)d_in[5];
  float* out = (float*)d_out;

  // workspace layout (bf16 counted in elements):
  //   qb  [8192,256] bf16   4 MB
  //   kb  [8192, 64] bf16   1 MB
  //   swg [8192,  4] f32  128 KB
  //   xb  [8192,1024] bf16 16 MB
  //   wqk [ 320,1024] bf16 640 KB
  unsigned short* qbuf = (unsigned short*)d_ws;
  unsigned short* kbuf = qbuf + (size_t)B_ * T_ * NH_ * DI_;
  float* swg           = (float*)(kbuf + (size_t)B_ * T_ * DI_);
  unsigned short* xb   = (unsigned short*)(swg + (size_t)B_ * T_ * NH_);
  unsigned short* wqk  = xb + (size_t)B_ * T_ * HID_;

  const int nX  = B_ * T_ * HID_;   // 8388608
  const int nWq = 256 * HID_;       // 262144
  const int nWk = 64 * HID_;        // 65536
  cvt_bf16_kernel<<<nX  / (8 * 256), 256, 0, stream>>>(x,  xb, nX);
  cvt_bf16_kernel<<<nWq / (8 * 256), 256, 0, stream>>>(Wq, wqk, nWq);
  cvt_bf16_kernel<<<nWk / (8 * 256), 256, 0, stream>>>(Wk, wqk + nWq, nWk);

  dim3 gA(5, (B_ * T_) / 128, 1);          // 320 cols / 64, 8192 rows / 128
  proj_qk_kernel<<<gA, 256, 0, stream>>>(xb, wqk, qbuf, kbuf);

  proj_w_kernel<<<(B_ * T_ * NH_) / 256, 256, 0, stream>>>(x, Ww, bw, swg);

  dim3 gC(T_ / 64, T_ / 128, B_);
  score_kernel<<<gC, 256, 0, stream>>>(qbuf, kbuf, swg, ib, out);
}
